// DetectionModel_54150947668455
// MI455X (gfx1250) — matrix operand order repaired and checked by reading, not compiled, not run on hardware
//
#include <hip/hip_runtime.h>
#include <hip/hip_bf16.h>
#include <math.h>

// ---------------------------------------------------------------------------
// CDNA5 (gfx1250) CenterPoint-style BEV detector.
//  - convs: implicit GEMM on v_wmma_f32_16x16x32_bf16
//  - input patches staged to LDS with the Tensor Data Mover (tensor_load_to_lds)
//  - activations halo-padded bf16 NHWC so the WMMA K-loop is branch-free
// ---------------------------------------------------------------------------

typedef __attribute__((ext_vector_type(16))) __bf16 v16bf;
typedef __attribute__((ext_vector_type(8)))  __bf16 v8bf;
typedef __attribute__((ext_vector_type(8)))  float  v8f;
typedef __attribute__((ext_vector_type(4)))  unsigned int v4u;
typedef __attribute__((ext_vector_type(8)))  int v8i;
typedef __attribute__((ext_vector_type(4)))  int v4i;
typedef __attribute__((address_space(3)))    __bf16 lds_bf16;

#define BEV_D 40
#define BEV_H 400
#define BEV_W 608
#define HW0   (BEV_H * BEV_W)

// ---------------------------------------------------------------------------
// Weight prep: fold BN scale into weights, emit bf16 in the WMMA B-fragment
// layout: idx = (((g*9+tap)*nKC + c)*32 + lane)*16 + e ;
// lanes 0-15 hold K=e, lanes 16-31 hold K=16+e (contiguous k order, same as the A fragment).
// ---------------------------------------------------------------------------
__global__ void prep_weights(const float* __restrict__ w,
                             const float* __restrict__ bg,
                             const float* __restrict__ bv,
                             __bf16* __restrict__ wb,
                             int CinR, int CinP, int CoutR, int CoutP)
{
    int idx   = blockIdx.x * blockDim.x + threadIdx.x;
    int total = 9 * CinP * CoutP;
    if (idx >= total) return;
    int e    = idx & 15;
    int lane = (idx >> 4) & 31;
    int rest = idx >> 9;
    int nKC  = CinP >> 5;
    int c    = rest % nKC;  rest /= nKC;
    int tap  = rest % 9;
    int gb   = rest / 9;
    int n     = lane & 15;
    int khalf = lane >> 4;
    int kk    = khalf * 16 + e;
    int cin   = c * 32 + kk;
    int cout  = gb * 16 + n;
    float val = 0.0f;
    if (cin < CinR && cout < CoutR) {
        float s = 1.0f;
        if (bg) s = bg[cout] * rsqrtf(bv[cout] + 1e-5f);
        val = w[((cout * CinR + cin) * 3 + tap / 3) * 3 + (tap % 3)] * s;
    }
    wb[idx] = (__bf16)val;
}

__global__ void prep_bias(const float* __restrict__ b,
                          const float* __restrict__ bg,
                          const float* __restrict__ bbt,
                          const float* __restrict__ bm,
                          const float* __restrict__ bv,
                          float* __restrict__ bias, int CoutR, int CoutP)
{
    int i = blockIdx.x * blockDim.x + threadIdx.x;
    if (i >= CoutP) return;
    float val = 0.0f;
    if (i < CoutR) {
        if (bg) {
            float s = bg[i] * rsqrtf(bv[i] + 1e-5f);
            val = (b[i] - bm[i]) * s + bbt[i];
        } else {
            val = b[i];
        }
    }
    bias[i] = val;
}

// f32 NCHW -> halo-padded bf16 NHWC (halo + pad channels zeroed).
__global__ void pack_nchw_bf16(const float* __restrict__ in,
                               __bf16* __restrict__ out,
                               int C, int Cp, int H, int W)
{
    int Wp = W + 2;
    int total = (H + 2) * Wp * Cp;
    int idx = blockIdx.x * blockDim.x + threadIdx.x;
    if (idx >= total) return;
    int c = idx % Cp;
    int t = idx / Cp;
    int x = t % Wp;
    int y = t / Wp;
    float v = 0.0f;
    if (y >= 1 && y <= H && x >= 1 && x <= W && c < C)
        v = in[c * H * W + (y - 1) * W + (x - 1)];
    out[idx] = (__bf16)v;
}

// Zero the 1-pixel halo ring of a padded NHWC buffer (run before each producer).
__global__ void zero_halo(__bf16* __restrict__ buf, int H, int W, int C)
{
    int Wp = W + 2;
    int cells = 2 * Wp + 2 * H;
    int idx = blockIdx.x * blockDim.x + threadIdx.x;
    if (idx >= cells * C) return;
    int c = idx % C;
    int t = idx / C;
    int y, x;
    if (t < Wp)          { y = 0;     x = t; }
    else if (t < 2 * Wp) { y = H + 1; x = t - Wp; }
    else { int t2 = t - 2 * Wp; y = 1 + (t2 >> 1); x = (t2 & 1) ? (W + 1) : 0; }
    buf[((size_t)y * Wp + x) * C + c] = (__bf16)0.0f;
}

// ---------------------------------------------------------------------------
// Implicit-GEMM 3x3 conv (pad=1) on WMMA bf16 -> f32.
// Block = 4 waves: shared 16-pixel tile, 4 x 16 output channels.
// Input patch (3 rows x span x Cin) staged into LDS by the Tensor Data Mover;
// A-fragments read from LDS, B-fragments streamed from L2 (pre-packed layout).
// Epilogue fuses bias(+BN), residual add, ReLU; output is padded bf16 NHWC.
// ---------------------------------------------------------------------------
__global__ void __launch_bounds__(128)
conv3x3_wmma(const __bf16* __restrict__ in, const __bf16* __restrict__ wb,
             const float* __restrict__ bias, const __bf16* __restrict__ resid,
             __bf16* __restrict__ out,
             int Hi, int Wi, int Cin, int Ho, int Wo, int Cout,
             int stride, int relu)
{
    __shared__ __bf16 smem[3 * 18 * 512];        // 55 KB max (stride2: 3*33*256 fits)
    const int span = 15 * stride + 3;            // 18 (s1) or 33 (s2) patch pixels
    const int Wip  = Wi + 2;
    const int oh   = blockIdx.y;
    const int ow0  = blockIdx.x * 16;
    const int lane = threadIdx.x & 31;
    const int wave = threadIdx.x >> 5;
    const int nKC  = Cin >> 5;

    // --- TDM: stage 3 x span x Cin bf16 patch into LDS (wave 0 issues) -------
    if (threadIdx.x < 32) {
        const __bf16* gsrc =
            in + ((size_t)(oh * stride) * Wip + (size_t)(ow0 * stride)) * Cin;
        unsigned long long ga = (unsigned long long)gsrc;
        unsigned la    = (unsigned)(unsigned long long)(lds_bf16*)smem;
        unsigned tile0 = (unsigned)(span * Cin);      // row length (2-byte elems)
        unsigned pitch = (unsigned)(Wip * Cin);       // tensor row pitch (elems)
        v4u g0;
        g0[0] = 1u;                                   // count=1, user descriptor
        g0[1] = la;                                   // lds_addr (bytes)
        g0[2] = (unsigned)ga;                         // global_addr[31:0]
        g0[3] = (unsigned)(ga >> 32) | (2u << 30);    // global_addr[56:32] | type=2
        v8i g1;
        g1[0] = (int)(1u << 16);                      // data_size=1 (2-byte elems)
        g1[1] = 0;                                    // tensor_dim0 = 0x4000_0000
        g1[2] = (int)0x4000u;                         // ..dim0 hi | tensor_dim1 lo
        g1[3] = (int)(0x4000u | (tile0 << 16));       // ..dim1 hi | tile_dim0
        g1[4] = 3;                                    // tile_dim1=3 rows, tile_dim2=0
        g1[5] = (int)pitch;                           // tensor_dim0_stride[31:0]
        g1[6] = 0;                                    // stride hi bits / dim1 stride
        g1[7] = 0;
        v4i gz4 = {0, 0, 0, 0};
        v8i gz8 = {0, 0, 0, 0, 0, 0, 0, 0};
        __builtin_amdgcn_tensor_load_to_lds(g0, g1, gz4, gz4, gz8, 0);
        __builtin_amdgcn_s_wait_tensorcnt(0);
    }
    __syncthreads();

    const int g = blockIdx.z * 4 + wave;          // 16-wide cout group
    if (g * 16 >= Cout) return;                   // wave-uniform (EXEC full for WMMA)

    const int l15 = lane & 15;
    const int kh  = lane >> 4;
    v8f acc = {};

    const v16bf* bbase = (const v16bf*)wb + (size_t)g * 9 * nKC * 32 + lane;

    for (int tap = 0; tap < 9; ++tap) {
        const int dy = tap / 3;                   // 0..2 patch row
        const int dx = tap % 3;                   // 0..2 patch col offset
        const __bf16* arow =
            &smem[((size_t)dy * span + (size_t)(l15 * stride + dx)) * Cin + kh * 16];
        const v16bf* bptr = bbase + (size_t)tap * nKC * 32;
        if (tap < 8)
            __builtin_prefetch((const void*)(bptr + (size_t)nKC * 32), 0, 1);
        for (int c = 0; c < nKC; ++c) {
            // A fragment (contiguous k order, as B): lanes 0-15 hold K {0..15},
            // lanes 16-31 hold K {16..31} -> two 16B ds loads.
            v8bf lo = *(const v8bf*)(arow + c * 32);
            v8bf hi = *(const v8bf*)(arow + c * 32 + 8);
            v16bf a;
#pragma unroll
            for (int e = 0; e < 8; ++e) { a[e] = lo[e]; a[e + 8] = hi[e]; }
            v16bf b = bptr[(size_t)c * 32];       // one contiguous 32B load
            acc = __builtin_amdgcn_wmma_f32_16x16x32_bf16(
                      false, a, false, b, (short)0, acc, false, false);
        }
    }

    // C/D layout: lane&15 = N (cout), row M = r + 8*(lane>>4).
    const int coutN = g * 16 + l15;
    const float bvv = bias[coutN];
    const int Wop = Wo + 2;
#pragma unroll
    for (int r = 0; r < 8; ++r) {
        const int owr = ow0 + r + kh * 8;
        if (owr < Wo) {
            float f = acc[r] + bvv;
            const size_t o = ((size_t)(oh + 1) * Wop + (owr + 1)) * Cout + coutN;
            if (resid) f += (float)resid[o];
            if (relu)  f = fmaxf(f, 0.0f);
            out[o] = (__bf16)f;
        }
    }
}

// ---------------------------------------------------------------------------
// Bilinear upsample on padded NHWC (align-corners linspace semantics).
// ---------------------------------------------------------------------------
__global__ void upsample_nhwc(const __bf16* __restrict__ in, __bf16* __restrict__ out,
                              int h, int w, int C, int Ho, int Wo)
{
    int idx = blockIdx.x * blockDim.x + threadIdx.x;
    if (idx >= Ho * Wo * C) return;
    int c  = idx % C;
    int t  = idx / C;
    int wo = t % Wo;
    int ho = t / Wo;
    float fy = (float)ho * (float)(h - 1) / (float)(Ho - 1);
    float fx = (float)wo * (float)(w - 1) / (float)(Wo - 1);
    int y0 = (int)floorf(fy); int y1 = min(y0 + 1, h - 1); float wy = fy - (float)y0;
    int x0 = (int)floorf(fx); int x1 = min(x0 + 1, w - 1); float wx = fx - (float)x0;
    int wp = w + 2;
    float v00 = (float)in[((size_t)(y0 + 1) * wp + (x0 + 1)) * C + c];
    float v01 = (float)in[((size_t)(y0 + 1) * wp + (x1 + 1)) * C + c];
    float v10 = (float)in[((size_t)(y1 + 1) * wp + (x0 + 1)) * C + c];
    float v11 = (float)in[((size_t)(y1 + 1) * wp + (x1 + 1)) * C + c];
    float top = v00 * (1.0f - wx) + v01 * wx;
    float bot = v10 * (1.0f - wx) + v11 * wx;
    out[((size_t)(ho + 1) * (Wo + 2) + (wo + 1)) * C + c] =
        (__bf16)(top * (1.0f - wy) + bot * wy);
}

// Head upsample x4: padded bf16 NHWC (h x w x Cp, first Co real) -> f32 NCHW chi.
__global__ void upsample_head(const __bf16* __restrict__ in, float* __restrict__ chi,
                              int h, int w, int Cp, int Co, int Ho, int Wo)
{
    int idx = blockIdx.x * blockDim.x + threadIdx.x;
    if (idx >= Co * Ho * Wo) return;
    int c  = idx / (Ho * Wo);
    int r  = idx % (Ho * Wo);
    int wo = r % Wo;
    int ho = r / Wo;
    float fy = (float)ho * (float)(h - 1) / (float)(Ho - 1);
    float fx = (float)wo * (float)(w - 1) / (float)(Wo - 1);
    int y0 = (int)floorf(fy); int y1 = min(y0 + 1, h - 1); float wy = fy - (float)y0;
    int x0 = (int)floorf(fx); int x1 = min(x0 + 1, w - 1); float wx = fx - (float)x0;
    int wp = w + 2;
    float v00 = (float)in[((size_t)(y0 + 1) * wp + (x0 + 1)) * Cp + c];
    float v01 = (float)in[((size_t)(y0 + 1) * wp + (x1 + 1)) * Cp + c];
    float v10 = (float)in[((size_t)(y1 + 1) * wp + (x0 + 1)) * Cp + c];
    float v11 = (float)in[((size_t)(y1 + 1) * wp + (x1 + 1)) * Cp + c];
    float top = v00 * (1.0f - wx) + v01 * wx;
    float bot = v10 * (1.0f - wx) + v11 * wx;
    chi[idx] = top * (1.0f - wy) + bot * wy;
}

__global__ void heat_kernel(const float* __restrict__ chi0, float* __restrict__ heat, int n)
{
    int i = blockIdx.x * blockDim.x + threadIdx.x;
    if (i < n) heat[i] = 1.0f / (1.0f + expf(-chi0[i]));
}

// 5x5 SAME max-pool NMS mask: score = (heat >= pooled) ? heat : -1
__global__ void score_kernel(const float* __restrict__ heat, float* __restrict__ scores,
                             int H, int W)
{
    int i = blockIdx.x * blockDim.x + threadIdx.x;
    if (i >= H * W) return;
    int x = i % W, y = i / W;
    float m = -1e30f;
    for (int dy = -2; dy <= 2; ++dy) {
        int yy = y + dy;
        if (yy < 0 || yy >= H) continue;
        for (int dx = -2; dx <= 2; ++dx) {
            int xx = x + dx;
            if (xx < 0 || xx >= W) continue;
            m = fmaxf(m, heat[yy * W + xx]);
        }
    }
    float h = heat[i];
    scores[i] = (h >= m) ? h : -1.0f;
}

// Exact top-k (stable, lower index wins ties): one block, k argmax sweeps.
__global__ void __launch_bounds__(1024)
topk_kernel(float* __restrict__ scores, float* __restrict__ tops,
            int* __restrict__ topi, int k, int total)
{
    __shared__ float sv[1024];
    __shared__ int   si[1024];
    int tid = threadIdx.x;
    for (int it = 0; it < k; ++it) {
        float best = -1e30f;
        int   bi   = 0x7fffffff;
        for (int i = tid; i < total; i += 1024) {
            float v = scores[i];
            if (v > best || (v == best && i < bi)) { best = v; bi = i; }
        }
        sv[tid] = best; si[tid] = bi;
        __syncthreads();
        for (int s = 512; s > 0; s >>= 1) {
            if (tid < s) {
                if (sv[tid + s] > sv[tid] ||
                    (sv[tid + s] == sv[tid] && si[tid + s] < si[tid])) {
                    sv[tid] = sv[tid + s]; si[tid] = si[tid + s];
                }
            }
            __syncthreads();
        }
        if (tid == 0) {
            tops[it] = sv[0];
            topi[it] = si[0];
            scores[si[0]] = -1e30f;   // consume (scores regenerated each launch)
        }
        __syncthreads();
    }
}

__global__ void decode_kernel(const float* __restrict__ chi,
                              const float* __restrict__ tops,
                              const int* __restrict__ topi,
                              float* __restrict__ out, int k)
{
    int t = blockIdx.x * blockDim.x + threadIdx.x;
    if (t >= k) return;
    const int S = BEV_H * BEV_W;
    int   i = topi[t];
    float s = tops[t];
    float row = (float)(i / BEV_W);
    float col = (float)(i % BEV_W);
    float offx = chi[1 * S + i];
    float offy = chi[2 * S + i];
    float xsz  = chi[3 * S + i];
    float ysz  = chi[4 * S + i];
    float st   = chi[5 * S + i];
    float ct   = chi[6 * S + i];
    float valid = (s > 0.05f) ? 1.0f : 0.0f;
    out[t * 6 + 0] = (offx + row) * valid;
    out[t * 6 + 1] = (offy + col) * valid;
    out[t * 6 + 2] = atan2f(st, ct) * valid;
    out[t * 6 + 3] = xsz * valid;
    out[t * 6 + 4] = ysz * valid;
    out[t * 6 + 5] = s * valid;
}

// ---------------------------------------------------------------------------
// Host orchestration
// ---------------------------------------------------------------------------
struct CD { int wI, bI, gI, CinR, CinP, CoutR, CoutP; };
static const CD g_cd[19] = {
    { 2, 3, 4,  40,  64,  32,  32},   //  0 c0   + bn0
    { 8, 9,-1,  32,  32,  64,  64},   //  1 c1   (stride 2)
    {10,11,12,  64,  64,  64,  64},   //  2 r1.c1+bn1
    {16,17,18,  64,  64,  64,  64},   //  3 r1.c2+bn2
    {22,23,-1,  64,  64, 128, 128},   //  4 c2   (stride 2)
    {24,25,26, 128, 128, 128, 128},   //  5 r2.c1
    {30,31,32, 128, 128, 128, 128},   //  6 r2.c2
    {36,37,-1, 128, 128, 256, 256},   //  7 c3   (stride 2)
    {38,39,40, 256, 256, 256, 256},   //  8 r3.c1
    {44,45,46, 256, 256, 256, 256},   //  9 r3.c2
    {50,51,-1, 256, 256, 512, 512},   // 10 c4   (stride 2)
    {52,53,54, 512, 512, 512, 512},   // 11 r4a.c1
    {58,59,60, 512, 512, 512, 512},   // 12 r4a.c2
    {64,65,66, 512, 512, 512, 512},   // 13 r4b.c1
    {70,71,72, 512, 512, 512, 512},   // 14 r4b.c2
    {76,77,-1, 512, 512, 256, 256},   // 15 c5
    {78,79,-1, 256, 256, 256, 256},   // 16 c6
    {80,81,82, 256, 256, 256, 256},   // 17 h0 + hbn
    {86,87,-1, 256, 256,   7,  16},   // 18 h1
};

extern "C" void kernel_launch(void* const* d_in, const int* in_sizes, int n_in,
                              void* d_out, int out_size, void* d_ws, size_t ws_size,
                              hipStream_t stream)
{
    (void)in_sizes; (void)n_in; (void)ws_size;
    char* ws = (char*)d_ws;
    size_t off = 0;
    auto alloc = [&](size_t bytes) -> char* {
        char* p = ws + off;
        off += (bytes + 255) & ~(size_t)255;
        return p;
    };

    const size_t SLACK = 65536;   // tail slack: TDM row overhang on right-edge tiles
    // Halo-padded activation buffers (bf16 NHWC), sized to max tensor per slot.
    __bf16* P0 = (__bf16*)alloc((size_t)402 * 610 * 64 * 2 + SLACK);
    __bf16* P1 = (__bf16*)alloc((size_t)402 * 610 * 32 * 2 + SLACK);
    __bf16* P2 = (__bf16*)alloc((size_t)202 * 306 * 64 * 2 + SLACK);
    float*  CHI    = (float*)alloc((size_t)7 * HW0 * 4);
    float*  HEAT   = (float*)alloc((size_t)HW0 * 4);
    float*  SCORES = (float*)alloc((size_t)HW0 * 4);
    float*  TOPS   = (float*)alloc(512);
    int*    TOPI   = (int*)alloc(512);

    // Packed weights + folded biases.
    __bf16* WB[19];
    float*  BI[19];
    for (int i = 0; i < 19; ++i) {
        WB[i] = (__bf16*)alloc((size_t)18 * g_cd[i].CinP * g_cd[i].CoutP);
        BI[i] = (float*)alloc((size_t)g_cd[i].CoutP * 4);
    }
    for (int i = 0; i < 19; ++i) {
        const CD& c = g_cd[i];
        const float* w   = (const float*)d_in[c.wI];
        const float* b   = (const float*)d_in[c.bI];
        const float* bg  = (c.gI >= 0) ? (const float*)d_in[c.gI]     : nullptr;
        const float* bbt = (c.gI >= 0) ? (const float*)d_in[c.gI + 1] : nullptr;
        const float* bm  = (c.gI >= 0) ? (const float*)d_in[c.gI + 2] : nullptr;
        const float* bv  = (c.gI >= 0) ? (const float*)d_in[c.gI + 3] : nullptr;
        int tot = 9 * c.CinP * c.CoutP;
        prep_weights<<<(tot + 255) / 256, 256, 0, stream>>>(w, bg, bv, WB[i],
                                                            c.CinR, c.CinP, c.CoutR, c.CoutP);
        prep_bias<<<(c.CoutP + 63) / 64, 64, 0, stream>>>(b, bg, bbt, bm, bv, BI[i],
                                                          c.CoutR, c.CoutP);
    }

    // Pack BEV input: f32 NCHW(40) -> padded bf16 NHWC(64), halo + pad zeroed.
    {
        int tot = 402 * 610 * 64;
        pack_nchw_bf16<<<(tot + 255) / 256, 256, 0, stream>>>(
            (const float*)d_in[0], P0, BEV_D, 64, BEV_H, BEV_W);
    }

    auto conv = [&](int li, const __bf16* in, __bf16* out, const __bf16* resid,
                    int Hi, int Wi, int stride, int relu) {
        int Cin  = g_cd[li].CinP;
        int Cout = g_cd[li].CoutP;
        int Ho = (stride == 1) ? Hi : ((Hi - 1) / 2 + 1);
        int Wo = (stride == 1) ? Wi : ((Wi - 1) / 2 + 1);
        int hcells = (2 * (Wo + 2) + 2 * Ho) * Cout;
        zero_halo<<<(hcells + 255) / 256, 256, 0, stream>>>(out, Ho, Wo, Cout);
        dim3 grid((Wo + 15) / 16, Ho, (Cout + 63) / 64);
        conv3x3_wmma<<<grid, 128, 0, stream>>>(in, WB[li], BI[li], resid, out,
                                               Hi, Wi, Cin, Ho, Wo, Cout, stride, relu);
    };
    auto up2 = [&](const __bf16* in, __bf16* out, int h, int w, int C) {
        int Ho = h * 2, Wo = w * 2;
        int hcells = (2 * (Wo + 2) + 2 * Ho) * C;
        zero_halo<<<(hcells + 255) / 256, 256, 0, stream>>>(out, Ho, Wo, C);
        int tot = Ho * Wo * C;
        upsample_nhwc<<<(tot + 255) / 256, 256, 0, stream>>>(in, out, h, w, C, Ho, Wo);
    };

    // Backbone
    conv(0,  P0, P1, nullptr, 400, 608, 1, 1);   // c0 + bn0 + relu
    conv(1,  P1, P2, nullptr, 400, 608, 2, 0);   // c1 stride2
    conv(2,  P2, P0, nullptr, 200, 304, 1, 1);   // r1.c1
    conv(3,  P0, P1, P2,      200, 304, 1, 1);   // r1.c2 + resid + relu
    conv(4,  P1, P2, nullptr, 200, 304, 2, 0);   // c2 stride2
    conv(5,  P2, P0, nullptr, 100, 152, 1, 1);   // r2.c1
    conv(6,  P0, P1, P2,      100, 152, 1, 1);   // r2.c2
    conv(7,  P1, P2, nullptr, 100, 152, 2, 0);   // c3 stride2
    conv(8,  P2, P0, nullptr,  50,  76, 1, 1);   // r3.c1
    conv(9,  P0, P1, P2,       50,  76, 1, 1);   // r3.c2
    conv(10, P1, P2, nullptr,  50,  76, 2, 0);   // c4 stride2
    conv(11, P2, P0, nullptr,  25,  38, 1, 1);   // r4a.c1
    conv(12, P0, P1, P2,       25,  38, 1, 1);   // r4a.c2
    conv(13, P1, P0, nullptr,  25,  38, 1, 1);   // r4b.c1
    conv(14, P0, P2, P1,       25,  38, 1, 1);   // r4b.c2

    // Neck: up2 -> c5 -> up2 -> c6
    up2(P2, P0, 25, 38, 512);
    conv(15, P0, P1, nullptr, 50, 76, 1, 0);     // c5
    up2(P1, P0, 50, 76, 256);
    conv(16, P0, P1, nullptr, 100, 152, 1, 0);   // c6
    conv(17, P1, P0, nullptr, 100, 152, 1, 1);   // h0 + hbn + relu
    conv(18, P0, P1, nullptr, 100, 152, 1, 0);   // h1 -> 7 (padded 16) channels

    // Head upsample x4 -> chi f32 [7][400][608]
    {
        int tot = 7 * HW0;
        upsample_head<<<(tot + 255) / 256, 256, 0, stream>>>(P1, CHI, 100, 152, 16, 7,
                                                             BEV_H, BEV_W);
    }

    // Decode
    heat_kernel<<<(HW0 + 255) / 256, 256, 0, stream>>>(CHI, HEAT, HW0);
    score_kernel<<<(HW0 + 255) / 256, 256, 0, stream>>>(HEAT, SCORES, BEV_H, BEV_W);
    int k = out_size / 6;
    if (k > 128) k = 128;
    topk_kernel<<<1, 1024, 0, stream>>>(SCORES, TOPS, TOPI, k, HW0);
    decode_kernel<<<(k + 127) / 128, 128, 0, stream>>>(CHI, TOPS, TOPI, (float*)d_out, k);
}
